// GNNRegressor_35433480192250
// MI455X (gfx1250) — matrix operand order repaired and checked by reading, not compiled, not run on hardware
//
#include <hip/hip_runtime.h>
#include <hip/hip_bf16.h>

typedef __attribute__((ext_vector_type(16))) _Float16 v16h;
typedef __attribute__((ext_vector_type(8)))  _Float16 v8h;
typedef __attribute__((ext_vector_type(4)))  _Float16 v4h;
typedef __attribute__((ext_vector_type(8)))  float    v8f;

#define HID 256

static __device__ __forceinline__ v16h hcat(v8h a, v8h b) {
  return __builtin_shufflevector(a, b, 0, 1, 2, 3, 4, 5, 6, 7, 8, 9, 10, 11, 12, 13, 14, 15);
}

// ---------------- elementwise helpers ----------------

__global__ void k_fill1(float* __restrict__ p, int n) {
  int i = blockIdx.x * blockDim.x + threadIdx.x;
  if (i < n) p[i] = 1.0f;
}

__global__ void k_zero4(float* __restrict__ p, int n4) {
  int i = blockIdx.x * blockDim.x + threadIdx.x;
  if (i < n4) ((float4*)p)[i] = make_float4(0.f, 0.f, 0.f, 0.f);
}

__global__ void k_degree(const int* __restrict__ ei, float* __restrict__ deg, int E) {
  int e = blockIdx.x * blockDim.x + threadIdx.x;
  if (e < E) atomicAdd(&deg[ei[E + e]], 1.0f);
}

__global__ void k_rsqrt(float* __restrict__ d, int n) {
  int i = blockIdx.x * blockDim.x + threadIdx.x;
  if (i < n) { float v = d[i]; d[i] = (v > 0.f) ? rsqrtf(v) : 0.f; }
}

// f32 -> f16 streaming convert (x -> h16)
__global__ void k_cvt16(const float* __restrict__ src, _Float16* __restrict__ dst, int n4) {
  int i = blockIdx.x * blockDim.x + threadIdx.x;
  if (i >= n4) return;
  float4 v = ((const float4*)src)[i];
  v4h o;
  o[0] = (_Float16)v.x; o[1] = (_Float16)v.y; o[2] = (_Float16)v.z; o[3] = (_Float16)v.w;
  ((v4h*)dst)[i] = o;
}

// WT[n*K + k] = (f16) W[k*256 + n]   (K x 256 row-major -> 256 x K f16)
__global__ void k_transpose_w(const float* __restrict__ W, _Float16* __restrict__ WT, int K) {
  int i = blockIdx.x * blockDim.x + threadIdx.x;
  if (i < HID * K) {
    int n = i / K, k = i - n * K;
    WT[i] = (_Float16)W[k * HID + n];
  }
}

// in-place f32 bias + relu (layer 2: feeds pooling)
__global__ void k_bias_relu(float* __restrict__ H, const float* __restrict__ b, int n4) {
  int i = blockIdx.x * blockDim.x + threadIdx.x;
  if (i >= n4) return;
  float4 v = ((float4*)H)[i];
  int fb = (i * 4) & (HID - 1);
  v.x = fmaxf(v.x + b[fb + 0], 0.f);
  v.y = fmaxf(v.y + b[fb + 1], 0.f);
  v.z = fmaxf(v.z + b[fb + 2], 0.f);
  v.w = fmaxf(v.w + b[fb + 3], 0.f);
  ((float4*)H)[i] = v;
}

// bias + relu, emitted as f16 A-operand for the next GEMM (layer 1: h1)
__global__ void k_bias_relu_f16(const float* __restrict__ H, const float* __restrict__ b,
                                _Float16* __restrict__ O, int n4) {
  int i = blockIdx.x * blockDim.x + threadIdx.x;
  if (i >= n4) return;
  float4 v = ((const float4*)H)[i];
  int fb = (i * 4) & (HID - 1);
  v4h o;
  o[0] = (_Float16)fmaxf(v.x + b[fb + 0], 0.f);
  o[1] = (_Float16)fmaxf(v.y + b[fb + 1], 0.f);
  o[2] = (_Float16)fmaxf(v.z + b[fb + 2], 0.f);
  o[3] = (_Float16)fmaxf(v.w + b[fb + 3], 0.f);
  ((v4h*)O)[i] = o;
}

// ---------------- WMMA GEMM: C[M,256] = A16[M,K] @ WT[256,K]^T ----------------
// Block = 128 threads (4 waves) per 16-row M tile. The 16xK f16 A strip is
// staged into LDS once per block with CDNA5 async-to-LDS loads (ASYNCcnt),
// then each wave computes a 16x64 strip: A via ds_load_b128 (bank-padded),
// B via two contiguous global_load_b128 from the pre-transposed f16 WT.
__global__ void k_gemm_wmma(const _Float16* __restrict__ A16, const _Float16* __restrict__ WT,
                            float* __restrict__ C, int M, int K) {
  __shared__ __align__(16) _Float16 Atile[16 * (HID + 8)];   // padded rows: K+8 halves
  const int tid  = threadIdx.x;
  const int lane = tid & 31;
  const int wv   = tid >> 5;
  const int hi   = lane >> 4;
  const int lo   = lane & 15;
  const int mtile = blockIdx.x;
  const int nbase = wv * 64;
  const int rs    = K + 8;                  // LDS row stride in halves

  // --- cooperative async stage of A rows [mtile*16, mtile*16+16) into LDS ---
  {
    const char* gsrc  = (const char*)(A16 + (size_t)mtile * 16 * K);
    uint32_t    lbase = (uint32_t)(uintptr_t)(&Atile[0]);
    const int   cpr   = K >> 3;             // 16B chunks per row
    const int   nch   = 16 * cpr;           // 256 (K=128) or 512 (K=256): multiple of 128
    for (int c = tid; c < nch; c += 128) {
      int r = c / cpr, j = c - r * cpr;
      uint64_t ga = (uint64_t)(uintptr_t)(gsrc + ((size_t)r * K + (size_t)j * 8) * 2);
      uint32_t la = lbase + (uint32_t)((r * rs + j * 8) * 2);
      asm volatile("global_load_async_to_lds_b128 %0, %1, off"
                   :: "v"(la), "v"(ga) : "memory");
    }
    asm volatile("s_wait_asynccnt 0x0" ::: "memory");
    __syncthreads();
  }

  v8f acc[4] = {v8f{}, v8f{}, v8f{}, v8f{}};

  for (int k0 = 0; k0 < K; k0 += 32) {
    // A 16x32 tile from LDS: lanes 0-15 rows 0-15 K[k0..k0+7]+[k0+16..k0+23],
    //                        lanes 16-31 rows 0-15 K[k0+8..]+[k0+24..]
    const _Float16* ap = &Atile[lo * rs + k0 + hi * 8];
    v16h a = hcat(*(const v8h*)ap, *(const v8h*)(ap + 16));
#pragma unroll
    for (int t = 0; t < 4; ++t) {
      // B 32x16 tile: lane (lane&15) = column n; K[k0+8h..+7]+[k0+16+8h..+7] from WT
      const _Float16* bp = WT + (size_t)(nbase + t * 16 + lo) * K + k0 + hi * 8;
      v16h b = hcat(*(const v8h*)bp, *(const v8h*)(bp + 16));
      acc[t] = __builtin_amdgcn_wmma_f32_16x16x32_f16(
          false, a, false, b, (short)0, acc[t], false, false);
    }
  }

  // C layout: lanes 0-15 -> rows +0..7, lanes 16-31 -> rows +8..15, col = lane&15
  const int mrow0 = mtile * 16 + hi * 8;
#pragma unroll
  for (int t = 0; t < 4; ++t) {
#pragma unroll
    for (int r = 0; r < 8; ++r) {
      int mr = mrow0 + r;
      if (mr < M) C[(size_t)mr * HID + (nbase + t * 16 + lo)] = acc[t][r];
    }
  }
}

// ---------------- sparse aggregation (gather * norm -> atomic scatter) ----------------
__global__ void k_aggregate(const float* __restrict__ H, float* __restrict__ OUT,
                            const int* __restrict__ ei, const float* __restrict__ dinv,
                            int E, int N) {
  int m = blockIdx.x * 8 + (threadIdx.x >> 5);
  if (m >= E + N) return;
  int lane = threadIdx.x & 31;
  int s, d;
  if (m < E) { s = ei[m]; d = ei[E + m]; }
  else       { s = m - E; d = s; }
  float nrm = dinv[s] * dinv[d];
  const float4* hp = (const float4*)(H + (size_t)s * HID);
  float* op = OUT + (size_t)d * HID;
#pragma unroll
  for (int c = 0; c < 2; ++c) {
    float4 v = hp[lane + c * 32];
    int base = (lane + c * 32) * 4;
    atomicAdd(op + base + 0, v.x * nrm);
    atomicAdd(op + base + 1, v.y * nrm);
    atomicAdd(op + base + 2, v.z * nrm);
    atomicAdd(op + base + 3, v.w * nrm);
  }
}

// ---------------- mean pool ----------------
__global__ void k_pool(const float* __restrict__ H, const int* __restrict__ batch,
                       float* __restrict__ pooled, float* __restrict__ counts, int N) {
  int node = blockIdx.x * 8 + (threadIdx.x >> 5);
  if (node >= N) return;
  int lane = threadIdx.x & 31;
  int g = batch[node];
  const float4* hp = (const float4*)(H + (size_t)node * HID);
  float* pp = pooled + (size_t)g * HID;
#pragma unroll
  for (int c = 0; c < 2; ++c) {
    float4 v = hp[lane + c * 32];
    int base = (lane + c * 32) * 4;
    atomicAdd(pp + base + 0, v.x);
    atomicAdd(pp + base + 1, v.y);
    atomicAdd(pp + base + 2, v.z);
    atomicAdd(pp + base + 3, v.w);
  }
  if (lane == 0) atomicAdd(&counts[g], 1.0f);
}

// ---------------- final: out[g] = (pooled[g]/max(cnt,1)) . Wfc + bfc ----------------
__global__ void k_final(const float* __restrict__ pooled, const float* __restrict__ counts,
                        const float* __restrict__ Wfc, const float* __restrict__ bfc,
                        float* __restrict__ out) {
  __shared__ float red[256];
  int g = blockIdx.x, c = threadIdx.x;
  float cnt = counts[g];
  cnt = cnt > 1.0f ? cnt : 1.0f;
  red[c] = pooled[(size_t)g * HID + c] / cnt * Wfc[c];
  __syncthreads();
  for (int s = 128; s > 0; s >>= 1) {
    if (c < s) red[c] += red[c + s];
    __syncthreads();
  }
  if (c == 0) out[g] = red[0] + bfc[0];
}

// ---------------- launch ----------------
extern "C" void kernel_launch(void* const* d_in, const int* in_sizes, int n_in,
                              void* d_out, int out_size, void* d_ws, size_t ws_size,
                              hipStream_t stream) {
  const float* x     = (const float*)d_in[0];
  const int*   ei    = (const int*)d_in[1];   // [2,E] flat: src then dst
  const int*   batch = (const int*)d_in[2];
  const float* W1    = (const float*)d_in[3];
  const float* b1    = (const float*)d_in[4];
  const float* W2    = (const float*)d_in[5];
  const float* b2    = (const float*)d_in[6];
  const float* Wfc   = (const float*)d_in[7];
  const float* bfc   = (const float*)d_in[8];
  float*       out   = (float*)d_out;
  (void)n_in; (void)ws_size;

  const int IN = 128;
  const int N  = in_sizes[0] / IN;   // 50000
  const int E  = in_sizes[1] / 2;    // 800000
  const int G  = out_size;           // 64

  // workspace carve-out (256B aligned)
  char* ws = (char*)d_ws;
  size_t off = 0;
  auto carve = [&](size_t bytes) -> void* {
    void* p = ws + off;
    off += (bytes + 255) & ~(size_t)255;
    return p;
  };
  float*     dinv   = (float*)carve((size_t)N * 4);
  float*     t0     = (float*)carve((size_t)N * HID * 4);
  float*     t1     = (float*)carve((size_t)N * HID * 4);
  _Float16*  h16    = (_Float16*)carve((size_t)N * HID * 2);  // f16 A operand
  _Float16*  wt1    = (_Float16*)carve((size_t)HID * IN * 2); // W1^T f16
  _Float16*  wt2    = (_Float16*)carve((size_t)HID * HID * 2);// W2^T f16
  float*     pooled = (float*)carve((size_t)G * HID * 4);
  float*     counts = (float*)carve((size_t)G * 4);           // contiguous after pooled

  const int NH4   = N * HID / 4;
  const int NI4   = N * IN / 4;
  const int MSG   = E + N;
  const int mtile = (N + 15) / 16;

  // symmetric normalization: deg = 1 (self loop) + in-degree; dinv = rsqrt(deg)
  k_fill1 <<<(N + 255) / 256, 256, 0, stream>>>(dinv, N);
  k_degree<<<(E + 255) / 256, 256, 0, stream>>>(ei, dinv, E);
  k_rsqrt <<<(N + 255) / 256, 256, 0, stream>>>(dinv, N);

  // one-time (per launch) operand prep: f16 x, f16 transposed weights
  k_cvt16      <<<(NI4 + 255) / 256, 256, 0, stream>>>(x, h16, NI4);
  k_transpose_w<<<(HID * IN  + 255) / 256, 256, 0, stream>>>(W1, wt1, IN);
  k_transpose_w<<<(HID * HID + 255) / 256, 256, 0, stream>>>(W2, wt2, HID);

  // layer 1: t0 = x @ W1 ; t1 = scatter(norm * gather(t0)) ; h16 = relu(t1 + b1) in f16
  k_gemm_wmma    <<<mtile, 128, 0, stream>>>(h16, wt1, t0, N, IN);
  k_zero4        <<<(NH4 + 255) / 256, 256, 0, stream>>>(t1, NH4);
  k_aggregate    <<<(MSG + 7) / 8, 256, 0, stream>>>(t0, t1, ei, dinv, E, N);
  k_bias_relu_f16<<<(NH4 + 255) / 256, 256, 0, stream>>>(t1, b1, h16, NH4);

  // layer 2: t0 = h1 @ W2 ; t1 = scatter(norm * gather(t0)) ; relu(t1 + b2) in f32
  k_gemm_wmma<<<mtile, 128, 0, stream>>>(h16, wt2, t0, N, HID);
  k_zero4    <<<(NH4 + 255) / 256, 256, 0, stream>>>(t1, NH4);
  k_aggregate<<<(MSG + 7) / 8, 256, 0, stream>>>(t0, t1, ei, dinv, E, N);
  k_bias_relu<<<(NH4 + 255) / 256, 256, 0, stream>>>(t1, b2, NH4);

  // mean pool + fc
  int pz4 = (G * HID + G) / 4;   // pooled + counts contiguous
  k_zero4<<<(pz4 + 255) / 256, 256, 0, stream>>>(pooled, pz4);
  k_pool <<<(N + 7) / 8, 256, 0, stream>>>(t1, batch, pooled, counts, N);
  k_final<<<G, 256, 0, stream>>>(pooled, counts, Wfc, bfc, out);
}
